// Random_Modality_GNN_25520695673364
// MI455X (gfx1250) — matrix operand order repaired and checked by reading, not compiled, not run on hardware
//
#include <hip/hip_runtime.h>
#include <hip/hip_bf16.h>

// ---------------------------------------------------------------------------
// Problem constants (match reference)
// ---------------------------------------------------------------------------
#define NN      50000
#define INF_    128
#define HH      128
#define OUTC    2
#define EE      800000
#define GG      32
#define EPS_BN  1e-5f

typedef __attribute__((ext_vector_type(16))) __bf16 v16bf;
typedef __attribute__((ext_vector_type(8)))  float  v8f;

// ---------------------------------------------------------------------------
// Weight prep: f32 [K x 128] row-major  ->  bf16 WMMA B-operand tiled layout.
// Tile (kt, nt): lane holds 16 contiguous bf16:
//   col = nt*16 + (lane&15), k = kt*32 + (lane>>4)*16 + j   (contiguous k order, same as the A loads)
// Linear index: ((nt*KT + kt)*32 + lane)*16 + j
// ---------------------------------------------------------------------------
__global__ __launch_bounds__(256) void k_prep_w(const float* __restrict__ W,
                                                __bf16* __restrict__ Wb, int K) {
  int idx = blockIdx.x * 256 + threadIdx.x;
  if (idx >= K * 128) return;
  int j    = idx & 15;
  int lane = (idx >> 4) & 31;
  int tile = idx >> 9;           // nt*KT + kt
  int KT   = K >> 5;
  int nt   = tile / KT;
  int kt   = tile - nt * KT;
  int col  = nt * 16 + (lane & 15);
  int k    = kt * 32 + (lane >> 4) * 16 + j;
  Wb[idx] = (__bf16)W[(size_t)k * 128 + col];
}

// ---------------------------------------------------------------------------
// WMMA GEMM: C[M x 128] = act(A[M x K] @ Wb + bias (+skip)), K in {128, 256}.
// 4 waves / block, each wave computes a 16-row x 128-col strip.
// Weights staged via LDS (64 KB), shared by the 4 waves.
// flags: bit0 = relu, bit1 = add skip (skip stride 128)
// ---------------------------------------------------------------------------
__global__ __launch_bounds__(128)
void k_gemm(const float* __restrict__ A, int lda,
            const __bf16* __restrict__ Wb, int K,
            const float* __restrict__ bias,
            const float* __restrict__ skip,
            float* __restrict__ C, int ldc, int coff,
            int M, int flags) {
  __shared__ __bf16 sW[256 * 128];            // 64 KB (max K = 256)
  // cooperative copy of swizzled weights into LDS (16B vectors)
  {
    int nvec = (K * 128) >> 3;                // 8 bf16 per uint4
    const uint4* g = (const uint4*)Wb;
    uint4* l = (uint4*)sW;
    for (int i = threadIdx.x; i < nvec; i += 128) l[i] = g[i];
  }
  __syncthreads();

  const int wave  = threadIdx.x >> 5;
  const int lane  = threadIdx.x & 31;
  const int rbase = blockIdx.x * 64 + wave * 16;
  int arow = rbase + (lane & 15);
  if (arow >= M) arow = M - 1;                // clamped load; stores predicated
  const int khalf = (lane >> 4) * 16;         // 0 or 16
  const int KT    = K >> 5;

  v8f acc[8] = {};

  for (int kt = 0; kt < KT; ++kt) {
    const float* ap = A + (size_t)arow * lda + kt * 32 + khalf;
    __builtin_prefetch(ap + 32, 0, 0);        // global_prefetch_b8 (next k-chunk)
    float4 f0 = *(const float4*)(ap + 0);
    float4 f1 = *(const float4*)(ap + 4);
    float4 f2 = *(const float4*)(ap + 8);
    float4 f3 = *(const float4*)(ap + 12);
    v16bf a;
    a[0] = (__bf16)f0.x;  a[1] = (__bf16)f0.y;  a[2]  = (__bf16)f0.z;  a[3]  = (__bf16)f0.w;
    a[4] = (__bf16)f1.x;  a[5] = (__bf16)f1.y;  a[6]  = (__bf16)f1.z;  a[7]  = (__bf16)f1.w;
    a[8] = (__bf16)f2.x;  a[9] = (__bf16)f2.y;  a[10] = (__bf16)f2.z;  a[11] = (__bf16)f2.w;
    a[12] = (__bf16)f3.x; a[13] = (__bf16)f3.y; a[14] = (__bf16)f3.z;  a[15] = (__bf16)f3.w;
#pragma unroll
    for (int nt = 0; nt < 8; ++nt) {
      v16bf b = *(const v16bf*)(sW + (((nt * KT + kt) * 32 + lane) << 4));
      acc[nt] = __builtin_amdgcn_wmma_f32_16x16x32_bf16(
          false, a, false, b, (short)0, acc[nt], false, false);
    }
  }

  // Epilogue: D layout — lane l, vgpr v -> row = (l>>4)*8 + v, col = l&15
  const int roff  = (lane >> 4) * 8;
  const int col_l = lane & 15;
#pragma unroll
  for (int nt = 0; nt < 8; ++nt) {
    int col  = nt * 16 + col_l;
    float bv = bias ? bias[col] : 0.f;
#pragma unroll
    for (int v = 0; v < 8; ++v) {
      int row = rbase + roff + v;
      if (row < M) {
        float val = acc[nt][v] + bv;
        if (flags & 2) val += skip[(size_t)row * 128 + col];
        if (flags & 1) val = fmaxf(val, 0.f);
        C[(size_t)row * ldc + coff + col] = val;
      }
    }
  }
}

// ---------------------------------------------------------------------------
// Elementwise / reduction helpers
// ---------------------------------------------------------------------------
__global__ void k_fill(float* p, float v, int n) {
  int i = blockIdx.x * 256 + threadIdx.x;
  if (i < n) p[i] = v;
}
__global__ void k_deg_acc(const int* __restrict__ dst, float* deg, int ne) {
  int i = blockIdx.x * 256 + threadIdx.x;
  if (i < ne) atomicAdd(&deg[dst[i]], 1.f);
}
__global__ void k_rsqrt_ip(float* p, int n) {
  int i = blockIdx.x * 256 + threadIdx.x;
  if (i < n) p[i] = rsqrtf(p[i]);
}

// GCN edge scatter: one wave per edge, float4 per lane (128 floats / edge)
__global__ __launch_bounds__(256)
void k_gcn_scatter(const int* __restrict__ src, const int* __restrict__ dst,
                   const float* __restrict__ dinv, const float* __restrict__ xw,
                   float* __restrict__ out, int ldc, int coff, int ne) {
  int e = blockIdx.x * 8 + (threadIdx.x >> 5);
  if (e >= ne) return;
  int lane = threadIdx.x & 31;
  int s = src[e], d = dst[e];
  float w = dinv[s] * dinv[d];
  float4 v = *(const float4*)(xw + (size_t)s * 128 + lane * 4);
  float* o = out + (size_t)d * ldc + coff + lane * 4;
  atomicAdd(o + 0, w * v.x);  atomicAdd(o + 1, w * v.y);
  atomicAdd(o + 2, w * v.z);  atomicAdd(o + 3, w * v.w);
}
__global__ void k_gcn_self(const float* __restrict__ dinv, const float* __restrict__ xw,
                           const float* __restrict__ bias, float* __restrict__ out,
                           int ldc, int coff, int n) {
  int i = blockIdx.x * 256 + threadIdx.x;
  if (i >= n * 128) return;
  int r = i >> 7, c = i & 127;
  out[(size_t)r * ldc + coff + c] += dinv[r] * dinv[r] * xw[i] + bias[c];
}

// matvec: one wave per row (dot of 128)
__global__ __launch_bounds__(256)
void k_matvec(const float* __restrict__ X, const float* __restrict__ a,
              float* __restrict__ out, int n) {
  int r = blockIdx.x * 8 + (threadIdx.x >> 5);
  if (r >= n) return;
  int lane = threadIdx.x & 31;
  float4 x = *(const float4*)(X + (size_t)r * 128 + lane * 4);
  float4 v = *(const float4*)(a + lane * 4);
  float p = x.x * v.x + x.y * v.y + x.z * v.z + x.w * v.w;
  for (int off = 16; off; off >>= 1) p += __shfl_down(p, off, 32);
  if (lane == 0) out[r] = p;
}

__device__ inline void atomicMaxF(float* addr, float val) {
  if (val >= 0.f) atomicMax((int*)addr, __float_as_int(val));
  else            atomicMin((unsigned int*)addr, __float_as_uint(val));
}
__global__ void k_gat_logit(const int* __restrict__ src, const int* __restrict__ dst,
                            const float* __restrict__ es, const float* __restrict__ ed,
                            float* __restrict__ ebuf, float* __restrict__ m, int ne) {
  int i = blockIdx.x * 256 + threadIdx.x;
  if (i >= ne) return;
  float e = es[src[i]] + ed[dst[i]];
  e = (e > 0.f) ? e : 0.2f * e;                // leaky_relu(0.2)
  ebuf[i] = e;
  atomicMaxF(&m[dst[i]], e);
}
__global__ void k_gat_exp(const int* __restrict__ dst, const float* __restrict__ m,
                          float* __restrict__ ebuf, float* __restrict__ s, int ne) {
  int i = blockIdx.x * 256 + threadIdx.x;
  if (i >= ne) return;
  float ex = __expf(ebuf[i] - m[dst[i]]);
  ebuf[i] = ex;
  atomicAdd(&s[dst[i]], ex);
}
__global__ __launch_bounds__(256)
void k_gat_scatter(const int* __restrict__ src, const int* __restrict__ dst,
                   const float* __restrict__ ebuf, const float* __restrict__ s,
                   const float* __restrict__ hs, float* __restrict__ out,
                   int ldc, int coff, int ne) {
  int e = blockIdx.x * 8 + (threadIdx.x >> 5);
  if (e >= ne) return;
  int lane = threadIdx.x & 31;
  int sn = src[e], d = dst[e];
  float alpha = ebuf[e] / (s[d] + 1e-16f);
  float4 v = *(const float4*)(hs + (size_t)sn * 128 + lane * 4);
  float* o = out + (size_t)d * ldc + coff + lane * 4;
  atomicAdd(o + 0, alpha * v.x);  atomicAdd(o + 1, alpha * v.y);
  atomicAdd(o + 2, alpha * v.z);  atomicAdd(o + 3, alpha * v.w);
}
__global__ void k_addbias(const float* __restrict__ bias, float* __restrict__ out,
                          int ldc, int coff, int n) {
  int i = blockIdx.x * 256 + threadIdx.x;
  if (i >= n * 128) return;
  int r = i >> 7, c = i & 127;
  out[(size_t)r * ldc + coff + c] += bias[c];
}

// batchnorm (training-mode, biased variance) — column stats then apply
__global__ __launch_bounds__(128)
void k_bn_stats(const float* __restrict__ X, float* __restrict__ sum,
                float* __restrict__ sq, int n) {
  int c = threadIdx.x;
  float s = 0.f, s2 = 0.f;
  for (int r = blockIdx.x; r < n; r += gridDim.x) {
    float v = X[(size_t)r * 128 + c];
    s += v;  s2 += v * v;
  }
  atomicAdd(&sum[c], s);
  atomicAdd(&sq[c], s2);
}
__global__ void k_bn_apply(float* __restrict__ X, const float* __restrict__ sum,
                           const float* __restrict__ sq, const float* __restrict__ g,
                           const float* __restrict__ be, int n, int relu) {
  int i = blockIdx.x * 256 + threadIdx.x;
  if (i >= n * 128) return;
  int c = i & 127;
  float inv = 1.f / (float)n;
  float mu  = sum[c] * inv;
  float var = sq[c] * inv - mu * mu;
  float v = g[c] * (X[i] - mu) * rsqrtf(var + EPS_BN) + be[c];
  X[i] = relu ? fmaxf(v, 0.f) : v;
}

// mean pool scatter
__global__ void k_pool(const float* __restrict__ X, const int* __restrict__ batch,
                       float* __restrict__ pooled, float* __restrict__ cnt, int n) {
  int i = blockIdx.x * 256 + threadIdx.x;
  if (i >= n * 128) return;
  int r = i >> 7, c = i & 127;
  int b = batch[r];
  atomicAdd(&pooled[(size_t)b * 128 + c], X[i]);
  if (c == 0) atomicAdd(&cnt[b], 1.f);
}

// final 3-layer MLP per graph per type, summed into out
__global__ __launch_bounds__(128)
void k_final(const float* __restrict__ pooled, const float* __restrict__ cnt,
             const float* __restrict__ W1, const float* __restrict__ b1,
             const float* __restrict__ W2, const float* __restrict__ b2,
             const float* __restrict__ W3, const float* __restrict__ b3,
             float* __restrict__ out) {
  __shared__ float rep[128], hid[128];
  int g = blockIdx.x, j = threadIdx.x;
  float c = fmaxf(cnt[g], 1.f);
  rep[j] = pooled[(size_t)g * 128 + j] / c;
  __syncthreads();
  float a = b1[j];
  for (int k = 0; k < 128; ++k) a += rep[k] * W1[k * 128 + j];
  hid[j] = fmaxf(a, 0.f);
  __syncthreads();
  a = b2[j];
  for (int k = 0; k < 128; ++k) a += hid[k] * W2[k * 128 + j];
  rep[j] = fmaxf(a, 0.f);
  __syncthreads();
  if (j < OUTC) {
    float o = b3[j];
    for (int k = 0; k < 128; ++k) o += rep[k] * W3[k * OUTC + j];
    atomicAdd(&out[g * OUTC + j], o);
  }
}

// ---------------------------------------------------------------------------
// Host orchestration
// ---------------------------------------------------------------------------
extern "C" void kernel_launch(void* const* d_in, const int* in_sizes, int n_in,
                              void* d_out, int out_size, void* d_ws, size_t ws_size,
                              hipStream_t stream) {
  (void)in_sizes; (void)n_in; (void)out_size; (void)ws_size;
  auto P = [&](int i) { return (const float*)d_in[i]; };

  // ---- input unpack (setup_inputs() dict order, params flattened in order) ----
  const float* x1 = P(0);
  const float* x2 = P(1);
  const int* ei11 = (const int*)d_in[2];
  const int* ei22 = (const int*)d_in[3];
  const int* ei12 = (const int*)d_in[4];
  const int* ei21 = (const int*)d_in[5];
  const int* batch1 = (const int*)d_in[6];
  const int* batch2 = (const int*)d_in[7];
  // params: pre[3][2](W,b,g,be)=24  convs[3]{gcn1:2,gcn2:2,gat12:5,gat21:5,cat1:2,cat2:2}=54
  //         post[3][2](W,b,g,be)=24  final[2][3](W,b)=12
  const int PRE0 = 8, CONV0 = 32, POST0 = 86, FIN0 = 110;

  // ---- workspace carve-out ----
  const size_t NF = (size_t)NN * 128;
  float* f   = (float*)d_ws;
  float* xA1 = f + 0 * NF;
  float* xA2 = f + 1 * NF;
  float* xB1 = f + 2 * NF;
  float* xB2 = f + 3 * NF;
  float* cat = f + 4 * NF;            // N x 256
  float* tmp = f + 6 * NF;            // xw / hs
  float* tm2 = f + 7 * NF;            // hd
  float* sc  = f + 8 * NF;
  float* dinv  = sc;
  float* esb   = sc + 1 * NN;
  float* edb   = sc + 2 * NN;
  float* mbuf  = sc + 3 * NN;
  float* sbuf  = sc + 4 * NN;
  float* ebuf  = sc + 5 * NN;         // E floats
  float* csum  = ebuf + EE;
  float* csq   = csum + 128;
  float* pooled = csq + 128;          // G*128
  float* cntb   = pooled + GG * 128;  // G
  __bf16* wb_cur = (__bf16*)(cntb + GG + 64);

  const size_t NB = NF * sizeof(float);

  // ---- helpers ----
  auto prep = [&](const float* W, int K) -> const __bf16* {
    __bf16* dst = wb_cur;
    int el = K * 128;
    hipLaunchKernelGGL(k_prep_w, dim3((el + 255) / 256), dim3(256), 0, stream, W, dst, K);
    wb_cur += el;
    return dst;
  };
  auto gemm = [&](const float* A, int lda, const __bf16* Wb, int K,
                  const float* bias, const float* skip,
                  float* C, int ldc, int coff, int flags) {
    hipLaunchKernelGGL(k_gemm, dim3((NN + 63) / 64), dim3(128), 0, stream,
                       A, lda, Wb, K, bias, skip, C, ldc, coff, NN, flags);
  };
  auto bnorm = [&](float* X, const float* g, const float* be, int relu) {
    hipMemsetAsync(csum, 0, 128 * sizeof(float), stream);
    hipMemsetAsync(csq, 0, 128 * sizeof(float), stream);
    hipLaunchKernelGGL(k_bn_stats, dim3(256), dim3(128), 0, stream, X, csum, csq, NN);
    hipLaunchKernelGGL(k_bn_apply, dim3((NN * 128 + 255) / 256), dim3(256), 0, stream,
                       X, csum, csq, g, be, NN, relu);
  };

  // ---- convert all GEMM weights to swizzled bf16 (deterministic per call) ----
  const __bf16 *wpre[2], *wgcn1[3], *wgcn2[3], *wg12s[3], *wg12d[3],
               *wg21s[3], *wg21d[3], *wcat1[3], *wcat2[3], *wpost[3][2];
  {
    int pb = PRE0 + 2 * 8;                   // only pre[2] survives
    wpre[0] = prep(P(pb + 0), 128);
    wpre[1] = prep(P(pb + 4), 128);
    for (int l = 0; l < 3; ++l) {
      int B = CONV0 + l * 18;
      wgcn1[l] = prep(P(B + 0), 128);
      wgcn2[l] = prep(P(B + 2), 128);
      wg12s[l] = prep(P(B + 4), 128);
      wg12d[l] = prep(P(B + 5), 128);
      wg21s[l] = prep(P(B + 9), 128);
      wg21d[l] = prep(P(B + 10), 128);
      wcat1[l] = prep(P(B + 14), 256);
      wcat2[l] = prep(P(B + 16), 256);
    }
    for (int i = 0; i < 3; ++i)
      for (int t = 0; t < 2; ++t)
        wpost[i][t] = prep(P(POST0 + i * 8 + t * 4), 128);
  }

  // ---- pre-processing (only last pre layer matters) ----
  {
    int pb = PRE0 + 2 * 8;
    gemm(x1, 128, wpre[0], 128, P(pb + 1), nullptr, xA1, 128, 0, 0);
    bnorm(xA1, P(pb + 2), P(pb + 3), 1);
    gemm(x2, 128, wpre[1], 128, P(pb + 5), nullptr, xA2, 128, 0, 0);
    bnorm(xA2, P(pb + 6), P(pb + 7), 1);
  }
  float* xc1 = xA1; float* xc2 = xA2;
  float* xn1 = xB1; float* xn2 = xB2;

  auto gcn = [&](const float* x, const __bf16* Wb, const float* bias,
                 const int* ei, int coff) {
    const int* esrc = ei; const int* edst = ei + EE;
    hipLaunchKernelGGL(k_fill, dim3((NN + 255) / 256), dim3(256), 0, stream, dinv, 1.f, NN);
    hipLaunchKernelGGL(k_deg_acc, dim3((EE + 255) / 256), dim3(256), 0, stream, edst, dinv, EE);
    hipLaunchKernelGGL(k_rsqrt_ip, dim3((NN + 255) / 256), dim3(256), 0, stream, dinv, NN);
    gemm(x, 128, Wb, 128, nullptr, nullptr, tmp, 128, 0, 0);
    hipLaunchKernelGGL(k_gcn_scatter, dim3((EE + 7) / 8), dim3(256), 0, stream,
                       esrc, edst, dinv, tmp, cat, 256, coff, EE);
    hipLaunchKernelGGL(k_gcn_self, dim3((NN * 128 + 255) / 256), dim3(256), 0, stream,
                       dinv, tmp, bias, cat, 256, coff, NN);
  };
  auto gat = [&](const float* xs, const float* xd, const __bf16* Ws, const __bf16* Wd,
                 const float* a_s, const float* a_d, const float* bias,
                 const int* ei, int coff) {
    const int* esrc = ei; const int* edst = ei + EE;
    gemm(xs, 128, Ws, 128, nullptr, nullptr, tmp, 128, 0, 0);   // hs
    gemm(xd, 128, Wd, 128, nullptr, nullptr, tm2, 128, 0, 0);   // hd
    hipLaunchKernelGGL(k_matvec, dim3((NN + 7) / 8), dim3(256), 0, stream, tmp, a_s, esb, NN);
    hipLaunchKernelGGL(k_matvec, dim3((NN + 7) / 8), dim3(256), 0, stream, tm2, a_d, edb, NN);
    hipLaunchKernelGGL(k_fill, dim3((NN + 255) / 256), dim3(256), 0, stream,
                       mbuf, -3.402823466e38f, NN);
    hipMemsetAsync(sbuf, 0, NN * sizeof(float), stream);
    hipLaunchKernelGGL(k_gat_logit, dim3((EE + 255) / 256), dim3(256), 0, stream,
                       esrc, edst, esb, edb, ebuf, mbuf, EE);
    hipLaunchKernelGGL(k_gat_exp, dim3((EE + 255) / 256), dim3(256), 0, stream,
                       edst, mbuf, ebuf, sbuf, EE);
    hipLaunchKernelGGL(k_gat_scatter, dim3((EE + 7) / 8), dim3(256), 0, stream,
                       esrc, edst, ebuf, sbuf, tmp, cat, 256, coff, EE);
    hipLaunchKernelGGL(k_addbias, dim3((NN * 128 + 255) / 256), dim3(256), 0, stream,
                       bias, cat, 256, coff, NN);
  };

  // ---- 3 message-passing layers ----
  for (int l = 0; l < 3; ++l) {
    int B = CONV0 + l * 18;
    // dst type graph_1: gcn(ei11) || gat21(ei21), then cat1 + skip + relu
    hipMemsetAsync(cat, 0, 2 * NB, stream);
    gcn(xc1, wgcn1[l], P(B + 1), ei11, 0);
    gat(xc2, xc1, wg21s[l], wg21d[l], P(B + 11), P(B + 12), P(B + 13), ei21, 128);
    gemm(cat, 256, wcat1[l], 256, P(B + 15), xc1, xn1, 128, 0, 3);
    // dst type graph_2: gcn(ei22) || gat12(ei12), then cat2 + skip + relu
    hipMemsetAsync(cat, 0, 2 * NB, stream);
    gcn(xc2, wgcn2[l], P(B + 3), ei22, 0);
    gat(xc1, xc2, wg12s[l], wg12d[l], P(B + 6), P(B + 7), P(B + 8), ei12, 128);
    gemm(cat, 256, wcat2[l], 256, P(B + 17), xc2, xn2, 128, 0, 3);
    float* t;
    t = xc1; xc1 = xn1; xn1 = t;
    t = xc2; xc2 = xn2; xn2 = t;
  }

  // ---- post-processing ----
  for (int i = 0; i < 3; ++i) {
    int relu = (i < 2) ? 1 : 0;
    int b0 = POST0 + i * 8;
    gemm(xc1, 128, wpost[i][0], 128, P(b0 + 1), nullptr, xn1, 128, 0, 0);
    bnorm(xn1, P(b0 + 2), P(b0 + 3), relu);
    gemm(xc2, 128, wpost[i][1], 128, P(b0 + 5), nullptr, xn2, 128, 0, 0);
    bnorm(xn2, P(b0 + 6), P(b0 + 7), relu);
    float* t;
    t = xc1; xc1 = xn1; xn1 = t;
    t = xc2; xc2 = xn2; xn2 = t;
  }

  // ---- pooling + final MLPs, summed over types ----
  hipMemsetAsync(d_out, 0, GG * OUTC * sizeof(float), stream);
  const float* xs[2]   = {xc1, xc2};
  const int*  bats[2]  = {batch1, batch2};
  for (int t = 0; t < 2; ++t) {
    hipMemsetAsync(pooled, 0, GG * 128 * sizeof(float), stream);
    hipMemsetAsync(cntb, 0, GG * sizeof(float), stream);
    hipLaunchKernelGGL(k_pool, dim3((NN * 128 + 255) / 256), dim3(256), 0, stream,
                       xs[t], bats[t], pooled, cntb, NN);
    int fb = FIN0 + t * 6;
    hipLaunchKernelGGL(k_final, dim3(GG), dim3(128), 0, stream,
                       pooled, cntb, P(fb), P(fb + 1), P(fb + 2), P(fb + 3),
                       P(fb + 4), P(fb + 5), (float*)d_out);
  }
}
